// DINRichVariant_62010737819846
// MI455X (gfx1250) — compile-verified, matrix operand order checked by reading, not run on hardware
//
#include <hip/hip_runtime.h>
#include <hip/hip_bf16.h>
#include <math.h>

// ---------------------------------------------------------------------------
// DIN-rich forward, fused per-batch-row kernel for gfx1250 (MI455X).
// Attention MLP (B*S = 409600 rows, 384->64->32->1) via v_wmma_f32_16x16x32_f16.
// A-fragments synthesized from packed-f16 registers (no div/branches);
// B-fragments pre-swizzled to fragment-ready LDS layout (1x32B load each).
// ---------------------------------------------------------------------------

typedef __attribute__((ext_vector_type(16))) _Float16 v16h;
typedef __attribute__((ext_vector_type(8)))  _Float16 v8h;
typedef __attribute__((ext_vector_type(8)))  float    v8f;

#define B_SZ   2048
#define S_LEN  200
#define S_PAD  208      // 13 chunks of 16 rows
#define TDIM   96
#define SEQ_STRIDE 104  // halves: 208B rows -> 16B aligned, conflict-free columns

// ---- dynamic LDS layout (bytes), all 16B aligned --------------------------
#define SM_W1H   0                         // swizzled W1 f16: 49152
#define SM_W2H   49152                     // swizzled W2 f16:  4096
#define SM_SEQ   53248                     // 208*104 f16 = 43264
#define SM_TGTH  96512                     // 96 f16 (pad 256)
#define SM_TGTF  96768                     // 96 f32 = 384
#define SM_MASK  97152                     // 208 f32 = 832
#define SM_SCORE 97984                     // 208 f32 = 832
#define SM_WGT   98816                     // 208 f32 = 832
#define SM_RED   99648                     // 256 f32 = 1024
#define SM_II    100672                    // 208 i32
#define SM_CI    101504
#define SM_BI    102336
#define SM_MLP   103168                    // 352 f32 = 1408
#define SM_O1    104576                    // 128 f32
#define SM_O2    105088                    // 64 f32 (pad to 105344)
#define SM_SCR   105344                    // 8 waves * 2048B scratch
#define SM_TOTAL 121728

__device__ __forceinline__ float prelu_f(float x, float a) {
    return x >= 0.f ? x : a * x;
}

__device__ __host__ __forceinline__ int kmap(int e, int hl) {
    return (e < 8) ? (hl * 8 + e) : (16 + hl * 8 + (e - 8));
}

// all 256 threads must call; returns broadcast result
__device__ __forceinline__ float blk_red(float v, float* red, int tid, int is_max) {
    red[tid] = v;
    __syncthreads();
    #pragma unroll
    for (int o = 128; o > 0; o >>= 1) {
        if (tid < o)
            red[tid] = is_max ? fmaxf(red[tid], red[tid + o]) : (red[tid] + red[tid + o]);
        __syncthreads();
    }
    float r = red[0];
    __syncthreads();
    return r;
}

// fp32 -> fp16 weight conversion + swizzle to fragment-ready layout:
//   w1s[((kt*4+nt)*32 + lane)*16 + e] = W1[kk*64 + n]
//   kk = kt*32 + kmap(e, lane>>4),  n = nt*16 + (lane&15)
__global__ __launch_bounds__(256) void din_prep(const float* __restrict__ W1,
                                                const float* __restrict__ W2,
                                                _Float16* __restrict__ w1s,
                                                _Float16* __restrict__ w2s) {
    int o = blockIdx.x * 256 + threadIdx.x;
    if (o < 384 * 64) {
        int e = o & 15, lane = (o >> 4) & 31, t = o >> 9;   // t = kt*4+nt
        int nt = t & 3, kt = t >> 2;
        int kk = kt * 32 + kmap(e, lane >> 4);
        int n  = nt * 16 + (lane & 15);
        w1s[o] = (_Float16)W1[kk * 64 + n];
    }
    if (o < 64 * 32) {
        int e = o & 15, lane = (o >> 4) & 31, t = o >> 9;   // t = kt*2+nt
        int nt = t & 1, kt = t >> 1;
        int kk = kt * 32 + kmap(e, lane >> 4);
        int n  = nt * 16 + (lane & 15);
        w2s[o] = (_Float16)W2[kk * 32 + n];
    }
}

__global__ __launch_bounds__(256) void din_main(
    const int* __restrict__ item_seq, const int* __restrict__ cat_seq,
    const int* __restrict__ brand_seq,
    const int* __restrict__ t_item, const int* __restrict__ t_cat,
    const int* __restrict__ t_brand, const int* __restrict__ user_id,
    const float* __restrict__ item_tbl, const float* __restrict__ user_tbl,
    const float* __restrict__ cat_tbl, const float* __restrict__ brand_tbl,
    const float* __restrict__ b1, const float* __restrict__ a1,
    const float* __restrict__ b2, const float* __restrict__ a2,
    const float* __restrict__ W3, const float* __restrict__ b3,
    const float* __restrict__ Wm1, const float* __restrict__ bm1,
    const float* __restrict__ am1,
    const float* __restrict__ Wm2, const float* __restrict__ bm2,
    const float* __restrict__ am2,
    const float* __restrict__ Wm3, const float* __restrict__ bm3,
    const _Float16* __restrict__ gw1s, const _Float16* __restrict__ gw2s,
    float* __restrict__ out) {

    extern __shared__ __align__(16) char smem[];
    _Float16* w1h   = (_Float16*)(smem + SM_W1H);
    _Float16* w2h   = (_Float16*)(smem + SM_W2H);
    _Float16* seqh  = (_Float16*)(smem + SM_SEQ);
    _Float16* tgth  = (_Float16*)(smem + SM_TGTH);
    float*    tgtf  = (float*)(smem + SM_TGTF);
    float*    maskf = (float*)(smem + SM_MASK);
    float*    scores= (float*)(smem + SM_SCORE);
    float*    wgt   = (float*)(smem + SM_WGT);
    float*    red   = (float*)(smem + SM_RED);
    int*      ii    = (int*)(smem + SM_II);
    int*      ci    = (int*)(smem + SM_CI);
    int*      bi    = (int*)(smem + SM_BI);
    float*    mlp_in= (float*)(smem + SM_MLP);
    float*    o1    = (float*)(smem + SM_O1);
    float*    o2    = (float*)(smem + SM_O2);

    const int b   = blockIdx.x;
    const int tid = threadIdx.x;

    const float a1v  = a1[0], a2v = a2[0], b3v = b3[0];
    const float am1v = am1[0], am2v = am2[0];

    // ---- phase 1: stage weights + indices + embeddings into LDS ------------
    {   // swizzled f16 weights via 128-bit copies
        const uint4* s1 = (const uint4*)gw1s;  uint4* d1 = (uint4*)w1h;
        #pragma unroll 4
        for (int i = tid; i < (384 * 64 * 2) / 16; i += 256) d1[i] = s1[i];
        const uint4* s2 = (const uint4*)gw2s;  uint4* d2 = (uint4*)w2h;
        for (int i = tid; i < (64 * 32 * 2) / 16; i += 256) d2[i] = s2[i];
    }
    {   // sequence indices + mask
        const int* isq = item_seq  + (size_t)b * S_LEN;
        const int* csq = cat_seq   + (size_t)b * S_LEN;
        const int* bsq = brand_seq + (size_t)b * S_LEN;
        for (int s = tid; s < S_PAD; s += 256) {
            int iv = 0, cv = 0, bv = 0;
            if (s < S_LEN) { iv = isq[s]; cv = csq[s]; bv = bsq[s]; }
            ii[s] = iv; ci[s] = cv; bi[s] = bv;
            maskf[s] = (s < S_LEN && iv > 0) ? 1.f : 0.f;
        }
        if (tid < TDIM) {   // target embedding
            float v;
            if (tid < 64)      v = item_tbl [(size_t)t_item [b] * 64 + tid];
            else if (tid < 80) v = cat_tbl  [(size_t)t_cat  [b] * 16 + (tid - 64)];
            else               v = brand_tbl[(size_t)t_brand[b] * 16 + (tid - 80)];
            tgtf[tid] = v;
            tgth[tid] = (_Float16)v;
        }
    }
    __syncthreads();
    // gather rich seq embeddings -> f16 LDS tile
    for (int e = tid; e < S_PAD * TDIM; e += 256) {
        int s = e / TDIM, d = e - s * TDIM;
        float v = 0.f;
        if (s < S_LEN) {
            if (d < 64)      v = item_tbl [(size_t)ii[s] * 64 + d];
            else if (d < 80) v = cat_tbl  [(size_t)ci[s] * 16 + (d - 64)];
            else             v = brand_tbl[(size_t)bi[s] * 16 + (d - 80)];
        }
        seqh[s * SEQ_STRIDE + d] = (_Float16)v;
    }
    __syncthreads();

    // ---- phase 2: attention MLP via WMMA, one 16-row chunk per wave --------
    const int wid  = tid >> 5;
    const int lane = tid & 31;
    const int mrow = lane & 15;
    const int hl   = lane >> 4;
    _Float16* hbuf = (_Float16*)(smem + SM_SCR + wid * 2048);  // 16x64 f16
    float*    sbuf = (float*)hbuf;                             // 16x32 f32 (reuse)

    for (int c = wid; c < 13; c += 8) {
        const _Float16* srow = seqh + (c * 16 + mrow) * SEQ_STRIDE;

        // This lane's share of the seq row + target: 6 x 16B vector LDS loads
        // each.  Chunk i covers d = (i>>1)*32 + (i&1)*16 + hl*8 + [0,8).
        v8h sseg[6], tseg[6];
        #pragma unroll
        for (int i = 0; i < 6; ++i) {
            const int off = (i >> 1) * 32 + (i & 1) * 16 + hl * 8;
            sseg[i] = *(const v8h*)(srow + off);
            tseg[i] = *(const v8h*)(tgth + off);
        }

        v8f acc[4] = {};
        #pragma unroll
        for (int kt = 0; kt < 12; ++kt) {
            const int seg = kt / 3;      // constant after unroll
            const int j   = kt - seg * 3;
            v8h lo, hi;
            if (seg == 0)      { lo = sseg[2*j];           hi = sseg[2*j+1]; }
            else if (seg == 1) { lo = tseg[2*j];           hi = tseg[2*j+1]; }
            else if (seg == 2) { lo = sseg[2*j]*tseg[2*j]; hi = sseg[2*j+1]*tseg[2*j+1]; }
            else               { lo = sseg[2*j]-tseg[2*j]; hi = sseg[2*j+1]-tseg[2*j+1]; }
            v16h afr = __builtin_shufflevector(lo, hi,
                         0,1,2,3,4,5,6,7,8,9,10,11,12,13,14,15);
            #pragma unroll
            for (int nt = 0; nt < 4; ++nt) {
                // fragment-ready swizzled weights: one 32B LDS read per lane
                v16h bfr = *(const v16h*)(w1h + ((kt * 4 + nt) * 32 + lane) * 16);
                acc[nt] = __builtin_amdgcn_wmma_f32_16x16x32_f16(
                    false, afr, false, bfr, (short)0, acc[nt], false, false);
            }
        }
        // bias + PReLU -> f16 h-buffer
        #pragma unroll
        for (int nt = 0; nt < 4; ++nt) {
            int col = nt * 16 + mrow;
            float bias = b1[col];
            #pragma unroll
            for (int v = 0; v < 8; ++v) {
                int r = v + 8 * hl;
                float x = acc[nt][v] + bias;
                hbuf[r * 64 + col] = (_Float16)(x >= 0.f ? x : a1v * x);
            }
        }
        asm volatile("s_wait_dscnt 0" ::: "memory");

        // GEMM2: [16x64] @ W2[64x32]; A fragments are contiguous 16B runs
        v16h a2f[2];
        #pragma unroll
        for (int kt = 0; kt < 2; ++kt) {
            v8h lo = *(const v8h*)(hbuf + mrow * 64 + kt * 32 + hl * 8);
            v8h hi = *(const v8h*)(hbuf + mrow * 64 + kt * 32 + 16 + hl * 8);
            a2f[kt] = __builtin_shufflevector(lo, hi,
                        0,1,2,3,4,5,6,7,8,9,10,11,12,13,14,15);
        }
        v8f c2[2] = {};
        #pragma unroll
        for (int kt = 0; kt < 2; ++kt)
            #pragma unroll
            for (int nt = 0; nt < 2; ++nt) {
                v16h bfr = *(const v16h*)(w2h + ((kt * 2 + nt) * 32 + lane) * 16);
                c2[nt] = __builtin_amdgcn_wmma_f32_16x16x32_f16(
                    false, a2f[kt], false, bfr, (short)0, c2[nt], false, false);
            }
        // bias + PReLU, fold W3 -> per-row score partials in f32 scratch
        #pragma unroll
        for (int nt = 0; nt < 2; ++nt) {
            int col = nt * 16 + mrow;
            float bias = b2[col];
            float w3v  = W3[col];
            #pragma unroll
            for (int v = 0; v < 8; ++v) {
                int r = v + 8 * hl;
                float x = c2[nt][v] + bias;
                x = x >= 0.f ? x : a2v * x;
                sbuf[r * 32 + col] = x * w3v;
            }
        }
        asm volatile("s_wait_dscnt 0" ::: "memory");
        if (lane < 16) {
            int sg = c * 16 + lane;
            float sc = b3v;
            #pragma unroll
            for (int col = 0; col < 32; ++col) sc += sbuf[lane * 32 + col];
            float tw = __expf(0.1f * (float)(sg - (S_LEN - 1)));
            scores[sg] = (maskf[sg] > 0.f) ? sc * tw : -1e9f;
        }
    }
    __syncthreads();

    // ---- phase 3: softmax + interest + masked mean -------------------------
    float pm = -1e30f;
    for (int s = tid; s < S_LEN; s += 256) pm = fmaxf(pm, scores[s]);
    float mx = blk_red(pm, red, tid, 1);

    float ps = 0.f;
    for (int s = tid; s < S_PAD; s += 256) {
        float e = (s < S_LEN) ? __expf(scores[s] - mx) : 0.f;
        wgt[s] = e;
        ps += e;
    }
    float sum = blk_red(ps, red, tid, 0);

    float pc = 0.f;
    for (int s = tid; s < S_LEN; s += 256) pc += maskf[s];
    float cnt = blk_red(pc, red, tid, 0);

    const float inv_sum = 1.f / sum;
    const float inv_cnt = 1.f / (cnt + 1e-8f);

    if (tid < TDIM) {
        float wacc = 0.f, macc = 0.f;
        for (int s = 0; s < S_LEN; ++s) {
            float e = (float)seqh[s * SEQ_STRIDE + tid];
            wacc += wgt[s] * e;
            macc += maskf[s] * e;
        }
        mlp_in[tid]            = wacc * inv_sum;   // interest
        mlp_in[TDIM + tid]     = tgtf[tid];        // target
        mlp_in[2 * TDIM + tid] = macc * inv_cnt;   // masked mean
    }
    if (tid < 64)
        mlp_in[3 * TDIM + tid] = user_tbl[(size_t)user_id[b] * 64 + tid];
    __syncthreads();

    // ---- phase 4: final MLP 352 -> 128 -> 64 -> 1 --------------------------
    if (tid < 128) {
        float acc = bm1[tid];
        for (int i = 0; i < 352; ++i) acc += mlp_in[i] * Wm1[i * 128 + tid];
        o1[tid] = prelu_f(acc, am1v);
    }
    __syncthreads();
    if (tid < 64) {
        float acc = bm2[tid];
        for (int i = 0; i < 128; ++i) acc += o1[i] * Wm2[i * 64 + tid];
        o2[tid] = prelu_f(acc, am2v);
    }
    __syncthreads();
    if (tid == 0) {
        float acc = bm3[0];
        for (int i = 0; i < 64; ++i) acc += o2[i] * Wm3[i];
        out[b] = acc;
    }
}

extern "C" void kernel_launch(void* const* d_in, const int* in_sizes, int n_in,
                              void* d_out, int out_size, void* d_ws, size_t ws_size,
                              hipStream_t stream) {
    (void)in_sizes; (void)n_in; (void)out_size; (void)ws_size;
    const int*   item_seq  = (const int*)d_in[0];
    const int*   cat_seq   = (const int*)d_in[1];
    const int*   brand_seq = (const int*)d_in[2];
    const int*   t_item    = (const int*)d_in[3];
    const int*   t_cat     = (const int*)d_in[4];
    const int*   t_brand   = (const int*)d_in[5];
    const int*   user_id   = (const int*)d_in[6];
    const float* item_tbl  = (const float*)d_in[7];
    const float* user_tbl  = (const float*)d_in[8];
    const float* cat_tbl   = (const float*)d_in[9];
    const float* brand_tbl = (const float*)d_in[10];
    const float* W1  = (const float*)d_in[11];
    const float* b1  = (const float*)d_in[12];
    const float* a1  = (const float*)d_in[13];
    const float* W2  = (const float*)d_in[14];
    const float* b2  = (const float*)d_in[15];
    const float* a2  = (const float*)d_in[16];
    const float* W3  = (const float*)d_in[17];
    const float* b3  = (const float*)d_in[18];
    const float* Wm1 = (const float*)d_in[19];
    const float* bm1 = (const float*)d_in[20];
    const float* am1 = (const float*)d_in[21];
    const float* Wm2 = (const float*)d_in[22];
    const float* bm2 = (const float*)d_in[23];
    const float* am2 = (const float*)d_in[24];
    const float* Wm3 = (const float*)d_in[25];
    const float* bm3 = (const float*)d_in[26];
    float* out = (float*)d_out;

    _Float16* w1s = (_Float16*)d_ws;
    _Float16* w2s = (_Float16*)((char*)d_ws + 384 * 64 * 2);

    din_prep<<<(384 * 64 + 255) / 256, 256, 0, stream>>>(W1, W2, w1s, w2s);

    const int smem_bytes = SM_TOTAL;
    hipFuncSetAttribute(reinterpret_cast<const void*>(din_main),
                        hipFuncAttributeMaxDynamicSharedMemorySize, smem_bytes);
    din_main<<<B_SZ, 256, smem_bytes, stream>>>(
        item_seq, cat_seq, brand_seq, t_item, t_cat, t_brand, user_id,
        item_tbl, user_tbl, cat_tbl, brand_tbl,
        b1, a1, b2, a2, W3, b3,
        Wm1, bm1, am1, Wm2, bm2, am2, Wm3, bm3,
        w1s, w2s, out);
}
